// SchNetInteraction_4372276707778
// MI455X (gfx1250) — matrix operand order repaired and checked by reading, not compiled, not run on hardware
//
#include <hip/hip_runtime.h>
#include <hip/hip_bf16.h>

// ---------------------------------------------------------------------------
// Problem constants (from reference): B=8, A=1024, Nn=64, F=128, S=64
// ---------------------------------------------------------------------------
#define BATCH 8
#define NATOM 1024
#define NNBR  64
#define FDIM  128
#define SDIM  64
#define CUTOFF 5.0f
#define LN2F_   0.6931471805599453f
#define LOG2EF_ 1.4426950408889634f

typedef __attribute__((ext_vector_type(16))) __bf16 v16bf;
typedef __attribute__((ext_vector_type(8)))  float  v8f;

union Frag16 { v16bf v; uint4 q[2]; };

// float -> bf16 (round to nearest even), returned as raw u16 bits
static __device__ __forceinline__ unsigned short f2bf(float f) {
    unsigned u = __builtin_bit_cast(unsigned, f);
    unsigned r = u + 0x7FFFu + ((u >> 16) & 1u);
    return (unsigned short)(r >> 16);
}

// shifted softplus: softplus(x) - log(2), entirely in base-2 hardware ops.
//   t = 2^(-log2e*|x|) = e^-|x|, t in (0,1] -> 1+t in [1,2]: no denorm/inf
//   ssp = max(x,0) + ln2*(log2(1+t) - 1)
static __device__ __forceinline__ float sspf(float x) {
    float t = __builtin_amdgcn_exp2f(-LOG2EF_ * fabsf(x));   // v_exp_f32
    float l = __builtin_amdgcn_logf(1.0f + t);               // v_log_f32 (log2)
    return fmaxf(x, 0.0f) + LN2F_ * (l - 1.0f);
}

// ---------------------------------------------------------------------------
// WMMA fragment loads per CDNA5 ISA 7.12.2 (wave32)
//
// A (16x32 bf16, MxK): lane L -> M = L&15, half = L>>4.
//   a[0..7]  = K = kbase + half*8 + (0..7)          (VGPRs 0-3)
//   a[8..15] = K = kbase + 16 + half*8 + (0..7)     (VGPRs 4-7)
// Two contiguous 16B chunks from a row-major (M x ldk) bf16 array.
//
// B (32x16 bf16, KxN): lane L -> N = L&15, half = L>>4.
//   b[0..7]  = K = kbase + half*8 + (0..7), b[8..15] = K = kbase + 16 + half*8 + (0..7)
// Loaded from N-major (N x Ktot) pre-transposed bf16 weights.
//
// C/D (16x16 f32): lane L -> N = L&15; c[r] -> M = (L>>4)*8 + r.
// ---------------------------------------------------------------------------
static __device__ __forceinline__ v16bf load_afrag_lds(const unsigned short* sA,
                                                       int ldk, int m, int kbase,
                                                       int half) {
    Frag16 f;
    const unsigned short* rp = sA + m * ldk + kbase + half * 8;
    f.q[0] = *(const uint4*)(rp);
    f.q[1] = *(const uint4*)(rp + 16);
    return f.v;
}

static __device__ __forceinline__ v16bf load_bfrag_gl(const unsigned short* __restrict__ Wt,
                                                      int Ktot, int n, int kbase,
                                                      int half) {
    Frag16 f;
    const uint4* p = (const uint4*)(Wt + (size_t)n * Ktot + kbase + half * 8);
    f.q[0] = p[0];
    f.q[1] = p[2];
    return f.v;
}

// ---------------------------------------------------------------------------
// Kernel 0: dst[n*K + k] = bf16(src[k*N + n])   (N-major transposed weights)
// ---------------------------------------------------------------------------
__global__ void transpose_to_bf16(const float* __restrict__ src,
                                  unsigned short* __restrict__ dst,
                                  int K, int N) {
    int idx = blockIdx.x * 256 + threadIdx.x;
    if (idx < K * N) {
        int n = idx / K;
        int k = idx - n * K;
        dst[idx] = f2bf(src[(size_t)k * N + n]);
    }
}

// ---------------------------------------------------------------------------
// Kernel 1: Yout = Xin @ W (no bias/act), M rows in blocks of 64, K=N=128.
// One workgroup = 64 rows, 8 waves, each wave owns a 16-col strip.
// ---------------------------------------------------------------------------
__global__ __launch_bounds__(256) void rowgemm_kernel(
    const float* __restrict__ Xin, const unsigned short* __restrict__ Wt,
    float* __restrict__ Yout) {
    __shared__ unsigned short sA[64 * FDIM];

    const int m0g = blockIdx.x * 64;
    const float4* xin4 = (const float4*)(Xin + (size_t)m0g * FDIM);
    for (int i = threadIdx.x; i < 64 * FDIM / 4; i += 256) {
        float4 v = xin4[i];
        int o = i * 4;
        sA[o + 0] = f2bf(v.x); sA[o + 1] = f2bf(v.y);
        sA[o + 2] = f2bf(v.z); sA[o + 3] = f2bf(v.w);
    }
    __syncthreads();

    const int lane = threadIdx.x & 31;
    const int wave = threadIdx.x >> 5;
    const int half = lane >> 4;
    const int n    = wave * 16 + (lane & 15);

    v16bf bfr[4];
#pragma unroll
    for (int kk = 0; kk < 4; ++kk) bfr[kk] = load_bfrag_gl(Wt, FDIM, n, kk * 32, half);

#pragma unroll
    for (int mt = 0; mt < 4; ++mt) {
        v8f c;
#pragma unroll
        for (int i = 0; i < 8; ++i) c[i] = 0.0f;
#pragma unroll
        for (int kk = 0; kk < 4; ++kk) {
            v16bf a = load_afrag_lds(sA, FDIM, mt * 16 + (lane & 15), kk * 32, half);
            c = __builtin_amdgcn_wmma_f32_16x16x32_bf16(false, a, false, bfr[kk],
                                                        (short)0, c, false, false);
        }
#pragma unroll
        for (int r = 0; r < 8; ++r)
            Yout[(size_t)(m0g + mt * 16 + half * 8 + r) * FDIM + n] = c[r];
    }
}

// ---------------------------------------------------------------------------
// Kernel 2: fused cfconv. One workgroup per (b,a) pair.
//   H = ssp(f_ij(64x64) @ W1 + b1)        GEMM1, WMMA bf16
//   W = H(64x128) @ W2 + b2               GEMM2, WMMA bf16 (kept in regs!)
//   agg[f] = sum_n W[n,f]*cut(r)*mask * y[neighbors[n], f]
// ---------------------------------------------------------------------------
__global__ __launch_bounds__(256) void conv_kernel(
    const float* __restrict__ f_ij, const float* __restrict__ r_ij,
    const int* __restrict__ neighbors, const unsigned char* __restrict__ nbmask,
    const unsigned short* __restrict__ W1t, const float* __restrict__ b1,
    const unsigned short* __restrict__ W2t, const float* __restrict__ b2,
    const float* __restrict__ y, float* __restrict__ agg) {
    __shared__ unsigned short sF[NNBR * SDIM];   // 8 KB
    __shared__ unsigned short sH[NNBR * FDIM];   // 16 KB
    __shared__ float sScale[NNBR];
    __shared__ int   sNbr[NNBR];
    __shared__ float sPart[256];

    const int wg = blockIdx.x;          // b*A + a
    const int bb = wg >> 10;            // / NATOM

    // stage f_ij tile (64x64 f32 -> bf16)
    const float4* fin4 = (const float4*)(f_ij + (size_t)wg * (NNBR * SDIM));
    for (int i = threadIdx.x; i < NNBR * SDIM / 4; i += 256) {
        float4 v = fin4[i];
        int o = i * 4;
        sF[o + 0] = f2bf(v.x); sF[o + 1] = f2bf(v.y);
        sF[o + 2] = f2bf(v.z); sF[o + 3] = f2bf(v.w);
    }
    if (threadIdx.x < NNBR) {
        int j = wg * NNBR + threadIdx.x;
        float cmask = (r_ij[j] <= CUTOFF) ? 1.0f : 0.0f;
        sScale[threadIdx.x] = cmask * (nbmask[j] ? 1.0f : 0.0f);
        sNbr[threadIdx.x]   = neighbors[j];
    }
    __syncthreads();

    const int lane = threadIdx.x & 31;
    const int wave = threadIdx.x >> 5;
    const int half = lane >> 4;
    const int nn   = wave * 16 + (lane & 15);

    // ---- GEMM1: H = ssp(f_ij @ W1 + b1), K = 64 ----
    {
        const float b1n = b1[nn];
        v16bf bfr[2];
#pragma unroll
        for (int kk = 0; kk < 2; ++kk) bfr[kk] = load_bfrag_gl(W1t, SDIM, nn, kk * 32, half);
#pragma unroll
        for (int mt = 0; mt < 4; ++mt) {
            v8f c;
#pragma unroll
            for (int i = 0; i < 8; ++i) c[i] = b1n;
#pragma unroll
            for (int kk = 0; kk < 2; ++kk) {
                v16bf a = load_afrag_lds(sF, SDIM, mt * 16 + (lane & 15), kk * 32, half);
                c = __builtin_amdgcn_wmma_f32_16x16x32_bf16(false, a, false, bfr[kk],
                                                            (short)0, c, false, false);
            }
#pragma unroll
            for (int r = 0; r < 8; ++r)
                sH[(mt * 16 + half * 8 + r) * FDIM + nn] = f2bf(sspf(c[r]));
        }
    }
    __syncthreads();

    // ---- GEMM2 + gather/scale/reduce epilogue, K = 128 ----
    const float b2n = b2[nn];
    const float* ybase = y + (size_t)bb * (NATOM * FDIM);
    float partial = 0.0f;
    {
        v16bf bfr[4];
#pragma unroll
        for (int kk = 0; kk < 4; ++kk) bfr[kk] = load_bfrag_gl(W2t, FDIM, nn, kk * 32, half);
#pragma unroll
        for (int mt = 0; mt < 4; ++mt) {
            v8f c;
#pragma unroll
            for (int i = 0; i < 8; ++i) c[i] = b2n;
#pragma unroll
            for (int kk = 0; kk < 4; ++kk) {
                v16bf a = load_afrag_lds(sH, FDIM, mt * 16 + (lane & 15), kk * 32, half);
                c = __builtin_amdgcn_wmma_f32_16x16x32_bf16(false, a, false, bfr[kk],
                                                            (short)0, c, false, false);
            }
#pragma unroll
            for (int r = 0; r < 8; ++r) {
                int row = mt * 16 + half * 8 + r;          // neighbor index
                float w = c[r] * sScale[row];
                partial = fmaf(w, ybase[(size_t)sNbr[row] * FDIM + nn], partial);
            }
        }
    }
    sPart[threadIdx.x] = partial;
    __syncthreads();
    if (lane < 16)
        agg[(size_t)wg * FDIM + nn] =
            sPart[wave * 32 + lane] + sPart[wave * 32 + 16 + lane];
}

// ---------------------------------------------------------------------------
// Kernel 3: out = ssp(agg @ W_f2out + b_f2out) @ W_dense + b_dense
// ---------------------------------------------------------------------------
__global__ __launch_bounds__(256) void tail_kernel(
    const float* __restrict__ agg,
    const unsigned short* __restrict__ Wf2t, const float* __restrict__ b_f2out,
    const unsigned short* __restrict__ Wdt,  const float* __restrict__ b_dense,
    float* __restrict__ out) {
    __shared__ unsigned short sA[64 * FDIM];
    __shared__ unsigned short sV[64 * FDIM];

    const int m0g = blockIdx.x * 64;
    const float4* in4 = (const float4*)(agg + (size_t)m0g * FDIM);
    for (int i = threadIdx.x; i < 64 * FDIM / 4; i += 256) {
        float4 v = in4[i];
        int o = i * 4;
        sA[o + 0] = f2bf(v.x); sA[o + 1] = f2bf(v.y);
        sA[o + 2] = f2bf(v.z); sA[o + 3] = f2bf(v.w);
    }
    __syncthreads();

    const int lane = threadIdx.x & 31;
    const int wave = threadIdx.x >> 5;
    const int half = lane >> 4;
    const int n    = wave * 16 + (lane & 15);

    // stage 1: v = ssp(agg @ W_f2out + b)
    {
        const float bn1 = b_f2out[n];
        v16bf bfr[4];
#pragma unroll
        for (int kk = 0; kk < 4; ++kk) bfr[kk] = load_bfrag_gl(Wf2t, FDIM, n, kk * 32, half);
#pragma unroll
        for (int mt = 0; mt < 4; ++mt) {
            v8f c;
#pragma unroll
            for (int i = 0; i < 8; ++i) c[i] = bn1;
#pragma unroll
            for (int kk = 0; kk < 4; ++kk) {
                v16bf a = load_afrag_lds(sA, FDIM, mt * 16 + (lane & 15), kk * 32, half);
                c = __builtin_amdgcn_wmma_f32_16x16x32_bf16(false, a, false, bfr[kk],
                                                            (short)0, c, false, false);
            }
#pragma unroll
            for (int r = 0; r < 8; ++r)
                sV[(mt * 16 + half * 8 + r) * FDIM + n] = f2bf(sspf(c[r]));
        }
    }
    __syncthreads();

    // stage 2: out = v @ W_dense + b
    {
        const float bn2 = b_dense[n];
        v16bf bfr[4];
#pragma unroll
        for (int kk = 0; kk < 4; ++kk) bfr[kk] = load_bfrag_gl(Wdt, FDIM, n, kk * 32, half);
#pragma unroll
        for (int mt = 0; mt < 4; ++mt) {
            v8f c;
#pragma unroll
            for (int i = 0; i < 8; ++i) c[i] = bn2;
#pragma unroll
            for (int kk = 0; kk < 4; ++kk) {
                v16bf a = load_afrag_lds(sV, FDIM, mt * 16 + (lane & 15), kk * 32, half);
                c = __builtin_amdgcn_wmma_f32_16x16x32_bf16(false, a, false, bfr[kk],
                                                            (short)0, c, false, false);
            }
#pragma unroll
            for (int r = 0; r < 8; ++r)
                out[(size_t)(m0g + mt * 16 + half * 8 + r) * FDIM + n] = c[r];
        }
    }
}

// ---------------------------------------------------------------------------
// Host launcher
// inputs: 0 x, 1 r_ij, 2 f_ij, 3 neighbors, 4 neighbor_mask, 5 W_in2f,
//         6 W1, 7 b1, 8 W2, 9 b2, 10 W_f2out, 11 b_f2out, 12 W_dense, 13 b_dense
// ---------------------------------------------------------------------------
extern "C" void kernel_launch(void* const* d_in, const int* in_sizes, int n_in,
                              void* d_out, int out_size, void* d_ws, size_t ws_size,
                              hipStream_t stream) {
    const float*         x         = (const float*)d_in[0];
    const float*         r_ij      = (const float*)d_in[1];
    const float*         f_ij      = (const float*)d_in[2];
    const int*           neighbors = (const int*)d_in[3];
    const unsigned char* nbmask    = (const unsigned char*)d_in[4];
    const float*         W_in2f    = (const float*)d_in[5];
    const float*         W1        = (const float*)d_in[6];
    const float*         b1        = (const float*)d_in[7];
    const float*         W2        = (const float*)d_in[8];
    const float*         b2        = (const float*)d_in[9];
    const float*         W_f2out   = (const float*)d_in[10];
    const float*         b_f2out   = (const float*)d_in[11];
    const float*         W_dense   = (const float*)d_in[12];
    const float*         b_dense   = (const float*)d_in[13];
    float*               out       = (float*)d_out;

    // workspace layout (bytes)
    char* ws = (char*)d_ws;
    unsigned short* W1t  = (unsigned short*)(ws + 0);        // 128x64  bf16 = 16384 B
    unsigned short* W2t  = (unsigned short*)(ws + 16384);    // 128x128 bf16 = 32768 B
    unsigned short* Wint = (unsigned short*)(ws + 49152);    // 128x128 bf16
    unsigned short* Wf2t = (unsigned short*)(ws + 81920);    // 128x128 bf16
    unsigned short* Wdt  = (unsigned short*)(ws + 114688);   // 128x128 bf16
    float* y   = (float*)(ws + 147456);                      // 8*1024*128 f32 = 4 MB
    float* agg = (float*)(ws + 147456 + 4194304);            // 4 MB

    // 0) transpose + convert weights to N-major bf16
    transpose_to_bf16<<<(SDIM * FDIM + 255) / 256, 256, 0, stream>>>(W1, W1t, SDIM, FDIM);
    transpose_to_bf16<<<(FDIM * FDIM + 255) / 256, 256, 0, stream>>>(W2, W2t, FDIM, FDIM);
    transpose_to_bf16<<<(FDIM * FDIM + 255) / 256, 256, 0, stream>>>(W_in2f, Wint, FDIM, FDIM);
    transpose_to_bf16<<<(FDIM * FDIM + 255) / 256, 256, 0, stream>>>(W_f2out, Wf2t, FDIM, FDIM);
    transpose_to_bf16<<<(FDIM * FDIM + 255) / 256, 256, 0, stream>>>(W_dense, Wdt, FDIM, FDIM);

    const int M = BATCH * NATOM;  // 8192 rows

    // 1) y = x @ W_in2f (no bias, no activation)
    rowgemm_kernel<<<M / 64, 256, 0, stream>>>(x, Wint, y);

    // 2) fused filter-net + cfconv -> agg
    conv_kernel<<<BATCH * NATOM, 256, 0, stream>>>(f_ij, r_ij, neighbors, nbmask,
                                                   W1t, b1, W2t, b2, y, agg);

    // 3) out = ssp(agg @ W_f2out + b) @ W_dense + b
    tail_kernel<<<M / 64, 256, 0, stream>>>(agg, Wf2t, b_f2out, Wdt, b_dense, out);
}
